// Block_32323923870072
// MI455X (gfx1250) — compile-verified, matrix operand order checked by reading, not run on hardware
//
#include <hip/hip_runtime.h>
#include <math.h>

// ---------------------------------------------------------------------------
// Transformer block for MI455X (gfx1250): bf16 WMMA + async-LDS double buffer.
//   D=1024, H=16, head_dim=64, B=4, T=2048  -> ~275 GFLOP, compute bound.
// ---------------------------------------------------------------------------

typedef __bf16 bf16;
typedef __attribute__((ext_vector_type(16))) __bf16 v16bf;
typedef __attribute__((ext_vector_type(8)))  __bf16 v8bf;
typedef __attribute__((ext_vector_type(4)))  __bf16 v4bf;
typedef __attribute__((ext_vector_type(8)))  float  v8f;
typedef __attribute__((ext_vector_type(4)))  int    v4i;

// async-to-LDS builtin takes (v4i AS1* src, v4i AS3* dst, int offset, int cpol)
typedef __attribute__((address_space(1))) v4i gv4i;   // global int4
typedef __attribute__((address_space(3))) v4i sv4i;   // LDS int4

#define D_MODEL 1024
#define N_HEAD  16
#define HEAD_D  64
#define T_SEQ   2048
#define B_SZ    4
#define NTOK    (B_SZ * T_SEQ)

#if __has_builtin(__builtin_amdgcn_global_load_async_to_lds_b128) && \
    __has_builtin(__builtin_amdgcn_s_wait_asynccnt)
#define ASYNC_OK 1
#endif

union Frag16 { v16bf v; v8bf h[2]; };

// Load one 16x32 bf16 operand fragment (A layout; B layout identical when B
// is stored transposed [N,K] row-major).  ISA 7.12.2: lane L holds row L%16;
// its 16 K values are two contiguous 8-elem chunks at k=(L/16)*8 and 16+(L/16)*8.
__device__ inline v16bf load_frag(const bf16* __restrict__ base, int ldm, int lane) {
    const bf16* p = base + (size_t)(lane & 15) * ldm + ((lane >> 4) << 3);
    Frag16 f;
    f.h[0] = *(const v8bf*)(p);
    f.h[1] = *(const v8bf*)(p + 16);
    return f.v;
}

__device__ inline v8f wmma_bf16(v16bf a, v16bf b, v8f c) {
    return __builtin_amdgcn_wmma_f32_16x16x32_bf16(
        false, a, false, b, (short)0, c, false, false);
}

// one 16-byte global -> LDS chunk: async (ASYNCcnt) when available
__device__ inline void stage_chunk(const bf16* g, bf16* l) {
#ifdef ASYNC_OK
    __builtin_amdgcn_global_load_async_to_lds_b128(
        (gv4i*)g, (sv4i*)l, 0, 0);
#else
    *(v8bf*)l = *(const v8bf*)g;
#endif
}
__device__ inline void async_wait0() {
#ifdef ASYNC_OK
    __builtin_amdgcn_s_wait_asynccnt(0);
#endif
}

// ---------------------------------------------------------------------------
// Weight transpose + f32 -> bf16 convert:  wt[n][k] = (bf16) w[k][n]
// ---------------------------------------------------------------------------
__global__ __launch_bounds__(256)
void transpose_to_bf16(const float* __restrict__ w, bf16* __restrict__ wt,
                       int K, int N) {
    __shared__ float tile[32][33];
    int kb = blockIdx.x * 32, nb = blockIdx.y * 32;
    int tx = threadIdx.x & 31, ty = threadIdx.x >> 5;
    #pragma unroll
    for (int r = ty; r < 32; r += 8)
        tile[r][tx] = w[(size_t)(kb + r) * N + nb + tx];
    __syncthreads();
    #pragma unroll
    for (int r = ty; r < 32; r += 8)
        wt[(size_t)(nb + r) * K + kb + tx] = (bf16)tile[tx][r];
}

// ---------------------------------------------------------------------------
// LayerNorm (f32 in) -> bf16 out.  One block (256 thr = 8 waves) per row.
// ---------------------------------------------------------------------------
__global__ __launch_bounds__(256)
void layernorm_bf16(const float* __restrict__ x, const float* __restrict__ g,
                    const float* __restrict__ b, bf16* __restrict__ out) {
    int row = blockIdx.x;
    float4 v = ((const float4*)(x + (size_t)row * D_MODEL))[threadIdx.x];
    float s  = v.x + v.y + v.z + v.w;
    float s2 = v.x*v.x + v.y*v.y + v.z*v.z + v.w*v.w;
    #pragma unroll
    for (int m = 16; m >= 1; m >>= 1) {
        s  += __shfl_xor(s,  m, 32);
        s2 += __shfl_xor(s2, m, 32);
    }
    __shared__ float ss[8], ss2[8];
    int wave = threadIdx.x >> 5;
    if ((threadIdx.x & 31) == 0) { ss[wave] = s; ss2[wave] = s2; }
    __syncthreads();
    s = 0.f; s2 = 0.f;
    #pragma unroll
    for (int w = 0; w < 8; w++) { s += ss[w]; s2 += ss2[w]; }
    float mu   = s * (1.0f / D_MODEL);
    float var  = s2 * (1.0f / D_MODEL) - mu * mu;
    float rstd = rsqrtf(var + 1e-5f);
    float4 gg = ((const float4*)g)[threadIdx.x];
    float4 bb = ((const float4*)b)[threadIdx.x];
    v4bf o;
    o[0] = (bf16)((v.x - mu) * rstd * gg.x + bb.x);
    o[1] = (bf16)((v.y - mu) * rstd * gg.y + bb.y);
    o[2] = (bf16)((v.z - mu) * rstd * gg.z + bb.z);
    o[3] = (bf16)((v.w - mu) * rstd * gg.w + bb.w);
    *(v4bf*)(out + (size_t)row * D_MODEL + threadIdx.x * 4) = o;
}

// ---------------------------------------------------------------------------
// WMMA GEMM:  C[M,N] = A[M,K](bf16,row) @ Bt[N,K](bf16,row)^T  (+epilogues)
// 256 thr = 8 waves (2M x 4N); wave tile 32x64; block tile 64x256; K step 32.
// Block tile staged in LDS (async double-buffer); LDS row stride 40 bf16
// (20 banks) => fragment ds_read_b128 are bank-conflict free.
// ---------------------------------------------------------------------------
template<bool BIAS, bool GELU, bool RESID, bool OUTBF>
__global__ __launch_bounds__(256)
void gemm_wmma(const bf16* __restrict__ A, int lda,
               const bf16* __restrict__ Bt, int ldb,
               const float* __restrict__ bias,
               const float* __restrict__ resid, int ldr,
               void* __restrict__ Cout, int ldc, int K) {
    __shared__ __align__(16) bf16 sA[2][64][40];
    __shared__ __align__(16) bf16 sB[2][256][40];

    int tid  = threadIdx.x;
    int lane = tid & 31;
    int wave = tid >> 5;
    int wm = wave & 1, wn = wave >> 1;
    int mB = blockIdx.x * 64;
    int nB = blockIdx.y * 256;

    // staging assignment: A tile 64x32 = 256 16B-chunks (1/thread),
    //                     B tile 256x32 = 1024 chunks (4/thread)
    int srow = tid >> 2;            // 0..63
    int schk = (tid & 3) * 8;       // element offset of 16B chunk
    const bf16* gA = A  + (size_t)(mB + srow) * lda + schk;
    const bf16* gB = Bt + (size_t)(nB + srow) * ldb + schk;

    v8f acc[2][4] = {};
    int nk = K >> 5;

    // fill buffer 0
    {
        stage_chunk(gA, &sA[0][srow][schk]);
        #pragma unroll
        for (int i = 0; i < 4; i++)
            stage_chunk(gB + (size_t)i * 64 * ldb, &sB[0][srow + i * 64][schk]);
    }

    for (int kt = 0; kt < nk; kt++) {
        async_wait0();
        __syncthreads();                 // fills visible, prior reads done
        if (kt + 1 < nk) {               // start next tile into other buffer
            int nb2 = (kt + 1) & 1, k0 = (kt + 1) << 5;
            stage_chunk(gA + k0, &sA[nb2][srow][schk]);
            #pragma unroll
            for (int i = 0; i < 4; i++)
                stage_chunk(gB + (size_t)i * 64 * ldb + k0,
                            &sB[nb2][srow + i * 64][schk]);
        }
        int buf = kt & 1;
        v16bf a[2], b[4];
        #pragma unroll
        for (int i = 0; i < 2; i++)
            a[i] = load_frag(&sA[buf][wm * 32 + i * 16][0], 40, lane);
        #pragma unroll
        for (int j = 0; j < 4; j++)
            b[j] = load_frag(&sB[buf][wn * 64 + j * 16][0], 40, lane);
        #pragma unroll
        for (int i = 0; i < 2; i++)
            #pragma unroll
            for (int j = 0; j < 4; j++)
                acc[i][j] = wmma_bf16(a[i], b[j], acc[i][j]);
    }

    int half = lane >> 4, ln = lane & 15;
    #pragma unroll
    for (int i = 0; i < 2; i++) {
        #pragma unroll
        for (int j = 0; j < 4; j++) {
            int col   = (nB + wn * 64) + j * 16 + ln;
            int rbase = (mB + wm * 32) + i * 16 + half * 8;
            float bv = BIAS ? bias[col] : 0.f;
            #pragma unroll
            for (int v = 0; v < 8; v++) {
                size_t row = (size_t)(rbase + v);
                float c = acc[i][j][v];
                if (BIAS) c += bv;
                if (GELU) {
                    float t = 0.7978845608028654f * (c + 0.044715f * c * c * c);
                    float e = __expf(2.0f * t);               // tanh via exp
                    c = 0.5f * c * (1.0f + (e - 1.0f) / (e + 1.0f));
                }
                if (RESID) c += resid[row * (size_t)ldr + col];
                if (OUTBF) ((bf16*)Cout)[row * (size_t)ldc + col] = (bf16)c;
                else       ((float*)Cout)[row * (size_t)ldc + col] = c;
            }
        }
    }
}

// ---------------------------------------------------------------------------
// Flash attention (causal).  Grid: (T/64, B*H); block 128 thr = 4 waves.
// Each wave owns 16 q rows; keys in tiles of 32.  K fragments for tile kt+1
// are prefetched before the softmax VALU section (latency hiding).
// ---------------------------------------------------------------------------
__global__ __launch_bounds__(128)
void attention_wmma(const bf16* __restrict__ qkv, bf16* __restrict__ y) {
    const int LD = 3 * D_MODEL;
    int lane = threadIdx.x & 31;
    int wave = threadIdx.x >> 5;
    int half = lane >> 4, ln = lane & 15;
    int qb = blockIdx.x;
    int bh = blockIdx.y;
    int bb = bh >> 4, hh = bh & 15;
    size_t base = (size_t)bb * T_SEQ * LD;
    const bf16* Q  = qkv + base + hh * HEAD_D;
    const bf16* Kp = qkv + base + D_MODEL + hh * HEAD_D;
    const bf16* Vp = qkv + base + 2 * D_MODEL + hh * HEAD_D;

    int qrow0 = qb * 64 + wave * 16;

    v16bf qa0 = load_frag(Q + (size_t)qrow0 * LD + 0,  LD, lane);
    v16bf qa1 = load_frag(Q + (size_t)qrow0 * LD + 32, LD, lane);

    v8f o[4] = {};
    float mrow[8], lrow[8];
    #pragma unroll
    for (int v = 0; v < 8; v++) { mrow[v] = -1e30f; lrow[v] = 0.f; }

    __shared__ __align__(16) bf16 Vt[HEAD_D][32];
    __shared__ __align__(16) bf16 Pb[4][16][32];

    v16bf kf[4];
    kf[0] = load_frag(Kp + 0,                    LD, lane);
    kf[1] = load_frag(Kp + 32,                   LD, lane);
    kf[2] = load_frag(Kp + (size_t)16 * LD + 0,  LD, lane);
    kf[3] = load_frag(Kp + (size_t)16 * LD + 32, LD, lane);

    int nkt = (qb * 64 + 63) / 32 + 1;
    for (int kt = 0; kt < nkt; kt++) {
        int kbase = kt * 32;

        __syncthreads();                 // previous Vt reads done
        for (int idx = threadIdx.x; idx < 32 * HEAD_D; idx += 128) {
            int kk = idx >> 6, hd = idx & 63;
            Vt[hd][kk] = Vp[(size_t)(kbase + kk) * LD + hd];
        }
        __syncthreads();

        // S = Q @ K^T  (two 16x16 f32 fragments)
        v8f s[2] = {};
        s[0] = wmma_bf16(qa0, kf[0], s[0]);
        s[0] = wmma_bf16(qa1, kf[1], s[0]);
        s[1] = wmma_bf16(qa0, kf[2], s[1]);
        s[1] = wmma_bf16(qa1, kf[3], s[1]);

        // prefetch next tile's K fragments; softmax VALU below hides latency
        if (kt + 1 < nkt) {
            int nkb = kbase + 32;
            kf[0] = load_frag(Kp + (size_t)(nkb +  0) * LD + 0,  LD, lane);
            kf[1] = load_frag(Kp + (size_t)(nkb +  0) * LD + 32, LD, lane);
            kf[2] = load_frag(Kp + (size_t)(nkb + 16) * LD + 0,  LD, lane);
            kf[3] = load_frag(Kp + (size_t)(nkb + 16) * LD + 32, LD, lane);
        }

        // scale + causal mask + online softmax (row = half*8 + v, col = ln)
        #pragma unroll
        for (int v = 0; v < 8; v++) {
            int qrow = qrow0 + half * 8 + v;
            float a0 = s[0][v] * 0.125f;
            float a1 = s[1][v] * 0.125f;
            if (kbase + ln      > qrow) a0 = -1e30f;
            if (kbase + 16 + ln > qrow) a1 = -1e30f;
            float mx = fmaxf(a0, a1);
            #pragma unroll
            for (int msk = 1; msk < 16; msk <<= 1)
                mx = fmaxf(mx, __shfl_xor(mx, msk, 32));
            float mnew  = fmaxf(mrow[v], mx);
            float alpha = __expf(mrow[v] - mnew);
            float p0 = __expf(a0 - mnew);
            float p1 = __expf(a1 - mnew);
            float psum = p0 + p1;
            #pragma unroll
            for (int msk = 1; msk < 16; msk <<= 1)
                psum += __shfl_xor(psum, msk, 32);
            lrow[v] = lrow[v] * alpha + psum;
            mrow[v] = mnew;
            s[0][v] = p0;
            s[1][v] = p1;
            #pragma unroll
            for (int j = 0; j < 4; j++) o[j][v] *= alpha;
        }

        // C-layout f32 P -> A-layout bf16 via per-wave LDS (DS ops in-order)
        #pragma unroll
        for (int v = 0; v < 8; v++) {
            int r = half * 8 + v;
            Pb[wave][r][ln]      = (bf16)s[0][v];
            Pb[wave][r][16 + ln] = (bf16)s[1][v];
        }
        v16bf pa = load_frag(&Pb[wave][0][0], 32, lane);

        #pragma unroll
        for (int j = 0; j < 4; j++) {
            v16bf vb = load_frag(&Vt[j * 16][0], 32, lane);
            o[j] = wmma_bf16(pa, vb, o[j]);
        }
    }

    #pragma unroll
    for (int j = 0; j < 4; j++) {
        #pragma unroll
        for (int v = 0; v < 8; v++) {
            int r = qrow0 + half * 8 + v;
            float val = o[j][v] / lrow[v];
            y[(size_t)(bb * T_SEQ + r) * D_MODEL + hh * HEAD_D + j * 16 + ln] =
                (bf16)val;
        }
    }
}

// ---------------------------------------------------------------------------
// Orchestration
// ---------------------------------------------------------------------------
static inline size_t align256(size_t x) { return (x + 255) & ~(size_t)255; }

extern "C" void kernel_launch(void* const* d_in, const int* in_sizes, int n_in,
                              void* d_out, int out_size, void* d_ws, size_t ws_size,
                              hipStream_t stream) {
    (void)in_sizes; (void)n_in; (void)out_size; (void)ws_size;
    const float* x      = (const float*)d_in[0];
    const float* w_attn = (const float*)d_in[1];
    const float* w_o    = (const float*)d_in[2];
    const float* ln1_g  = (const float*)d_in[3];
    const float* ln1_b  = (const float*)d_in[4];
    const float* ln2_g  = (const float*)d_in[5];
    const float* ln2_b  = (const float*)d_in[6];
    const float* w_fc   = (const float*)d_in[7];
    const float* b_fc   = (const float*)d_in[8];
    const float* w_proj = (const float*)d_in[9];
    const float* b_proj = (const float*)d_in[10];
    float* out = (float*)d_out;

    char* ws = (char*)d_ws;
    size_t off = 0;
    auto alloc = [&](size_t bytes) { char* p = ws + off; off = align256(off + bytes); return p; };
    bf16*  wtAttn = (bf16*)alloc((size_t)3 * D_MODEL * D_MODEL * 2);
    bf16*  wtO    = (bf16*)alloc((size_t)D_MODEL * D_MODEL * 2);
    bf16*  wtFc   = (bf16*)alloc((size_t)4 * D_MODEL * D_MODEL * 2);
    bf16*  wtProj = (bf16*)alloc((size_t)D_MODEL * 4 * D_MODEL * 2);
    bf16*  lnBuf  = (bf16*)alloc((size_t)NTOK * D_MODEL * 2);
    bf16*  qkvBuf = (bf16*)alloc((size_t)NTOK * 3 * D_MODEL * 2);
    bf16*  yBuf   = (bf16*)alloc((size_t)NTOK * D_MODEL * 2);
    float* x1     = (float*)alloc((size_t)NTOK * D_MODEL * 4);
    bf16*  hBuf   = (bf16*)alloc((size_t)NTOK * 4 * D_MODEL * 2);

    transpose_to_bf16<<<dim3(D_MODEL/32, 3*D_MODEL/32), 256, 0, stream>>>(w_attn, wtAttn, D_MODEL, 3*D_MODEL);
    transpose_to_bf16<<<dim3(D_MODEL/32, D_MODEL/32),   256, 0, stream>>>(w_o,    wtO,    D_MODEL, D_MODEL);
    transpose_to_bf16<<<dim3(D_MODEL/32, 4*D_MODEL/32), 256, 0, stream>>>(w_fc,   wtFc,   D_MODEL, 4*D_MODEL);
    transpose_to_bf16<<<dim3(4*D_MODEL/32, D_MODEL/32), 256, 0, stream>>>(w_proj, wtProj, 4*D_MODEL, D_MODEL);

    layernorm_bf16<<<NTOK, 256, 0, stream>>>(x, ln1_g, ln1_b, lnBuf);

    gemm_wmma<false,false,false,true><<<dim3(NTOK/64, 3*D_MODEL/256), 256, 0, stream>>>(
        lnBuf, D_MODEL, wtAttn, D_MODEL, nullptr, nullptr, 0, qkvBuf, 3*D_MODEL, D_MODEL);

    attention_wmma<<<dim3(T_SEQ/64, B_SZ*N_HEAD), 128, 0, stream>>>(qkvBuf, yBuf);

    gemm_wmma<false,false,true,false><<<dim3(NTOK/64, D_MODEL/256), 256, 0, stream>>>(
        yBuf, D_MODEL, wtO, D_MODEL, nullptr, x, D_MODEL, x1, D_MODEL, D_MODEL);

    layernorm_bf16<<<NTOK, 256, 0, stream>>>(x1, ln2_g, ln2_b, lnBuf);

    gemm_wmma<true,true,false,true><<<dim3(NTOK/64, 4*D_MODEL/256), 256, 0, stream>>>(
        lnBuf, D_MODEL, wtFc, D_MODEL, b_fc, nullptr, 0, hBuf, 4*D_MODEL, D_MODEL);

    gemm_wmma<true,false,true,false><<<dim3(NTOK/64, D_MODEL/256), 256, 0, stream>>>(
        hBuf, 4*D_MODEL, wtProj, 4*D_MODEL, b_proj, x1, D_MODEL, out, D_MODEL, 4*D_MODEL);
}
